// LSTM_21801253995253
// MI455X (gfx1250) — matrix operand order repaired and checked by reading, not compiled, not run on hardware
//
#include <hip/hip_runtime.h>

// ---------------------------------------------------------------------------
// Fused LSTM (28 steps, H=64, I=28) + linear head + softmax for gfx1250.
// One wave32 owns 16 batch rows end-to-end; GEMMs via v_wmma_f32_16x16x32_f16.
// Round 3: sigmoid via V_TANH_F32 identity (0.5*tanh(x/2)+0.5) to kill the
// IEEE v_div_scale/v_div_fmas sequences on the recurrent critical path.
// ---------------------------------------------------------------------------

typedef __attribute__((ext_vector_type(16))) _Float16 v16h;
typedef __attribute__((ext_vector_type(8)))  float    v8f;

#define T_SEQ 28
#define N_IN  28
#define HID   64
#define ROWS  16      // batch rows per wave
#define WPB   4       // waves per block
#define HSTR  88      // padded h-stage row stride in halves (16B-aligned, bank-spread)

union Frag16 { int4 q[2]; v16h h; };

__device__ __forceinline__ v16h ld_b_frag(const _Float16* p) {
    // B fragment: 16 contiguous halves per lane
    Frag16 f;
    f.q[0] = *(const int4*)(p);
    f.q[1] = *(const int4*)(p + 8);
    return f.h;
}

__device__ __forceinline__ v16h ld_a_frag(const _Float16* p) {
    // A fragment: elements 0..7 at K=base..base+7, elements 8..15 at K=base+16..base+23
    Frag16 f;
    f.q[0] = *(const int4*)(p);
    f.q[1] = *(const int4*)(p + 16);
    return f.h;
}

__device__ __forceinline__ v8f wmma_f16(v16h a, v16h b, v8f c) {
    // (neg_a, A, neg_b, B, c_mod, C, reuse_a, reuse_b)
    return __builtin_amdgcn_wmma_f32_16x16x32_f16(false, a, false, b, (short)0, c,
                                                  false, false);
}

__device__ __forceinline__ float rcp_(float x) {
#if __has_builtin(__builtin_amdgcn_rcpf)
    return __builtin_amdgcn_rcpf(x);         // v_rcp_f32
#else
    return 1.0f / x;
#endif
}

__device__ __forceinline__ float tanh_(float x) {
#if __has_builtin(__builtin_amdgcn_tanhf)
    return __builtin_amdgcn_tanhf(x);        // V_TANH_F32 on gfx1250
#else
    float e = __expf(2.0f * x);
    return (e - 1.0f) * rcp_(e + 1.0f);
#endif
}

__device__ __forceinline__ float sigm_(float x) {
#if __has_builtin(__builtin_amdgcn_tanhf)
    // sigmoid(x) = 0.5*tanh(x/2) + 0.5 : mul + v_tanh + fma, no divide
    return fmaf(0.5f, __builtin_amdgcn_tanhf(0.5f * x), 0.5f);
#else
    return rcp_(1.0f + __expf(-x));
#endif
}

__global__ __launch_bounds__(WPB * 32)
void lstm_fused_kernel(const float* __restrict__ x,
                       const float* __restrict__ W_ih,
                       const float* __restrict__ W_hh,
                       const float* __restrict__ b_ih,
                       const float* __restrict__ b_hh,
                       const float* __restrict__ W_out,
                       const float* __restrict__ b_out,
                       float* __restrict__ out)
{
    // Weight fragments pre-swizzled into WMMA B-operand lane layout:
    //   element e of lane L holds K = 8*(L/16) + (e&7) + 16*(e>>3) (+32*kstep), column N = tile*16 + L%16
    __shared__ __align__(16) _Float16 sWih[16][32][16];        // 16 KB
    __shared__ __align__(16) _Float16 sWhh[2][16][32][16];     // 32 KB
    __shared__ __align__(16) _Float16 sH[WPB][ROWS][HSTR];     // 11 KB  per-wave h staging
    __shared__ float sWout[10][64];                            // 2.5 KB
    __shared__ float sBout[10];

    const int tid  = threadIdx.x;
    const int lane = tid & 31;
    const int wv   = tid >> 5;
    const int half = lane >> 4;     // 0: lanes 0-15, 1: lanes 16-31
    const int ln   = lane & 15;

    // ---- one-time cooperative build of LDS weight fragments ----
    for (int idx = tid; idx < 16 * 32 * 16; idx += WPB * 32) {
        int e    = idx & 15;
        int l    = (idx >> 4) & 31;
        int tile = idx >> 9;
        int n    = tile * 16 + (l & 15);
        int k    = ((l >> 4) << 3) + (e & 7) + ((e >> 3) << 4); // 0..31 (pad K>=28)
        float v  = (k < N_IN) ? W_ih[n * N_IN + k] : 0.0f;
        sWih[tile][l][e] = (_Float16)v;
    }
    for (int idx = tid; idx < 2 * 16 * 32 * 16; idx += WPB * 32) {
        int e    = idx & 15;
        int l    = (idx >> 4) & 31;
        int tile = (idx >> 9) & 15;
        int ks   = idx >> 13;
        int n    = tile * 16 + (l & 15);
        int k    = ks * 32 + ((l >> 4) << 3) + (e & 7) + ((e >> 3) << 4); // 0..63
        sWhh[ks][tile][l][e] = (_Float16)W_hh[n * HID + k];
    }
    for (int idx = tid; idx < 10 * 64; idx += WPB * 32)
        sWout[idx >> 6][idx & 63] = W_out[idx];
    if (tid < 10) sBout[tid] = b_out[tid];

    // per-lane combined bias, column n = tile*16 + ln (same for both lane halves,
    // matching the f32 C/D layout where N = lane%16)
    float biasr[16];
    #pragma unroll
    for (int tile = 0; tile < 16; ++tile) {
        int n = tile * 16 + ln;
        biasr[tile] = b_ih[n] + b_hh[n];
    }

    __syncthreads();

    const int rowBase = (blockIdx.x * WPB + wv) * ROWS;
    const float* xw = x + (size_t)rowBase * T_SEQ * N_IN;

    // cell state c: D-layout, 4 hidden sub-tiles x 8 VGPRs
    v8f c[4];
    #pragma unroll
    for (int s = 0; s < 4; ++s)
        #pragma unroll
        for (int j = 0; j < 8; ++j) c[s][j] = 0.0f;

    for (int t = 0; t < T_SEQ; ++t) {
        // ---- x A-fragment: row M = ln, K = half*8+e (e<8) / 16+half*8+(e-8) ----
        v16h xa;
        {
            const float4* xq = (const float4*)(xw + ((size_t)ln * T_SEQ + t) * N_IN);
            float4 f0 = xq[half * 2];
            float4 f1 = xq[half * 2 + 1];
            float4 f2, f3;
            if (half == 0) { f2 = xq[4]; f3 = xq[5]; }
            else           { f2 = xq[6]; f3 = make_float4(0.f, 0.f, 0.f, 0.f); }
            xa[0]  = (_Float16)f0.x; xa[1]  = (_Float16)f0.y;
            xa[2]  = (_Float16)f0.z; xa[3]  = (_Float16)f0.w;
            xa[4]  = (_Float16)f1.x; xa[5]  = (_Float16)f1.y;
            xa[6]  = (_Float16)f1.z; xa[7]  = (_Float16)f1.w;
            xa[8]  = (_Float16)f2.x; xa[9]  = (_Float16)f2.y;
            xa[10] = (_Float16)f2.z; xa[11] = (_Float16)f2.w;
            xa[12] = (_Float16)f3.x; xa[13] = (_Float16)f3.y;
            xa[14] = (_Float16)f3.z; xa[15] = (_Float16)f3.w;
        }

        // ---- h A-fragments for both k-steps (t==0: values unused) ----
        const _Float16* hp = &sH[wv][ln][0];
        v16h ha0 = ld_a_frag(hp + half * 8);
        v16h ha1 = ld_a_frag(hp + 32 + half * 8);

        // ---- gates in 4 hidden sub-tiles; i,f,g,o tiles share lane mapping ----
        #pragma unroll
        for (int st = 0; st < 4; ++st) {
            // Batch ALL B-fragment loads for this group up front so the 12 WMMAs
            // drain the DS pipe with partial waits instead of load->wait0->wmma.
            v16h bw0 = ld_b_frag(&sWih[st     ][lane][0]);
            v16h bw1 = ld_b_frag(&sWih[st +  4][lane][0]);
            v16h bw2 = ld_b_frag(&sWih[st +  8][lane][0]);
            v16h bw3 = ld_b_frag(&sWih[st + 12][lane][0]);
            v16h bh0, bh1, bh2, bh3, bh4, bh5, bh6, bh7;
            if (t > 0) {
                bh0 = ld_b_frag(&sWhh[0][st     ][lane][0]);
                bh1 = ld_b_frag(&sWhh[0][st +  4][lane][0]);
                bh2 = ld_b_frag(&sWhh[0][st +  8][lane][0]);
                bh3 = ld_b_frag(&sWhh[0][st + 12][lane][0]);
                bh4 = ld_b_frag(&sWhh[1][st     ][lane][0]);
                bh5 = ld_b_frag(&sWhh[1][st +  4][lane][0]);
                bh6 = ld_b_frag(&sWhh[1][st +  8][lane][0]);
                bh7 = ld_b_frag(&sWhh[1][st + 12][lane][0]);
            }

            v8f ai, af, ag, ao;
            #pragma unroll
            for (int j = 0; j < 8; ++j) {
                ai[j] = biasr[st];
                af[j] = biasr[st + 4];
                ag[j] = biasr[st + 8];
                ao[j] = biasr[st + 12];
            }
            // input projection (K=32, padded)
            ai = wmma_f16(xa, bw0, ai);
            af = wmma_f16(xa, bw1, af);
            ag = wmma_f16(xa, bw2, ag);
            ao = wmma_f16(xa, bw3, ao);
            if (t > 0) {
                // recurrence, K = 64 -> 2 k-steps
                ai = wmma_f16(ha0, bh0, ai);
                af = wmma_f16(ha0, bh1, af);
                ag = wmma_f16(ha0, bh2, ag);
                ao = wmma_f16(ha0, bh3, ao);
                ai = wmma_f16(ha1, bh4, ai);
                af = wmma_f16(ha1, bh5, af);
                ag = wmma_f16(ha1, bh6, ag);
                ao = wmma_f16(ha1, bh7, ao);
            }
            // ---- LSTM cell, pure per-lane (D layout: M = half*8+j, N = st*16+ln) ----
            #pragma unroll
            for (int j = 0; j < 8; ++j) {
                float iv = sigm_(ai[j]);
                float fv = sigm_(af[j]);
                float gv = tanh_(ag[j]);
                float ov = sigm_(ao[j]);
                float cj = fv * c[st][j] + iv * gv;
                c[st][j] = cj;
                sH[wv][half * 8 + j][st * 16 + ln] = (_Float16)(ov * tanh_(cj));
            }
        }
    }

    // ---- head: logits = h_T @ W_outT + b_out, softmax; lanes 0..15 own a row ----
    if (lane < 16) {
        const int row = rowBase + lane;
        float logits[10];
        #pragma unroll
        for (int cls = 0; cls < 10; ++cls) {
            float s = sBout[cls];
            #pragma unroll 8
            for (int k = 0; k < HID; ++k)
                s += (float)sH[wv][lane][k] * sWout[cls][k];
            logits[cls] = s;
        }
        float m = logits[0];
        #pragma unroll
        for (int cls = 1; cls < 10; ++cls) m = fmaxf(m, logits[cls]);
        float e[10], sum = 0.0f;
        #pragma unroll
        for (int cls = 0; cls < 10; ++cls) { e[cls] = __expf(logits[cls] - m); sum += e[cls]; }
        float inv = rcp_(sum);
        #pragma unroll
        for (int cls = 0; cls < 10; ++cls)
            out[(size_t)row * 10 + cls] = e[cls] * inv;
    }
}

extern "C" void kernel_launch(void* const* d_in, const int* in_sizes, int n_in,
                              void* d_out, int out_size, void* d_ws, size_t ws_size,
                              hipStream_t stream) {
    const float* x    = (const float*)d_in[0];
    const float* Wih  = (const float*)d_in[1];
    const float* Whh  = (const float*)d_in[2];
    const float* bih  = (const float*)d_in[3];
    const float* bhh  = (const float*)d_in[4];
    const float* Wout = (const float*)d_in[5];
    const float* bout = (const float*)d_in[6];
    float* out = (float*)d_out;

    const int batch  = in_sizes[0] / (T_SEQ * N_IN);   // 32768
    const int blocks = batch / (WPB * ROWS);           // 512
    lstm_fused_kernel<<<blocks, WPB * 32, 0, stream>>>(x, Wih, Whh, bih, bhh,
                                                       Wout, bout, out);
}
